// GraphFusion_66288525246841
// MI455X (gfx1250) — compile-verified, matrix operand order checked by reading, not run on hardware
//
#include <hip/hip_runtime.h>
#include <hip/hip_bf16.h>

typedef __bf16 bf16;
typedef __attribute__((ext_vector_type(16))) __bf16 v16bf;
typedef __attribute__((ext_vector_type(8)))  float  v8f;

#define B_TOTAL 65536
#define HID     256
#define TS      16          // samples per workgroup
#define MROWS   (TS * 3)    // 48 node-rows per tile
#define KBIG    1024        // heads*HID

union Frag { v16bf v; uint4 u[2]; };   // 32B: one bf16 WMMA A/B operand per lane
union B4   { bf16 h[4]; uint2 u; };
union B8   { bf16 h[8]; uint4 u; };

// ---------------- prep kernels (tiny, run once per launch into d_ws) ----------------

// WrT[n][h*256+f] = bf16( W[f, h*256+n] )   (head-reshuffled, transposed, bf16)
__global__ void prep_wr(const float* __restrict__ W, bf16* __restrict__ wrT) {
  int n = blockIdx.x;               // 0..255 output column
  int t = threadIdx.x;              // 0..255
  for (int i = 0; i < 4; ++i) {
    int k = t + i * 256;            // k = h*256 + f
    int h = k >> 8, f = k & 255;
    wrT[n * 1024 + k] = (bf16)W[f * 1024 + h * 256 + n];
  }
}

// WoT[n][k] = bf16( Wo[k, n] )
__global__ void prep_wo(const float* __restrict__ Wo, bf16* __restrict__ woT) {
  int n = blockIdx.x, k = threadIdx.x;
  woT[n * 256 + k] = (bf16)Wo[k * 256 + n];
}

// pT[q][f] (bf16, 16 rows): q<4 -> (W a_src)[f,q] ; q in 4..7 -> (W a_dst)[f,q-4] ;
// rows 8..15 zero padding so pT is a full 16-column WMMA B-operand.
__global__ void prep_p(const float* __restrict__ W, const float* __restrict__ a_src,
                       const float* __restrict__ a_dst, bf16* __restrict__ pT) {
  int q = blockIdx.x;               // 0..15
  int f = threadIdx.x;              // 0..255
  if (q < 8) {
    int h = q & 3;
    const float* a = (q < 4) ? (a_src + h * 256) : (a_dst + h * 256);
    float acc = 0.f;
    for (int o = 0; o < 256; ++o) acc += W[f * 1024 + h * 256 + o] * a[o];
    pT[q * 256 + f] = (bf16)acc;
  } else {
    pT[q * 256 + f] = (bf16)0.f;
  }
}

// ---------------- fused GAT layer (device) ----------------

__device__ __forceinline__ void gat_layer(
    bf16 (*s_x)[256], bf16 (*s_a)[1024],
    float (*s_sc)[2][3][4], float (*s_al)[3][3][4],
    const bf16* __restrict__ pT, const bf16* __restrict__ wrT,
    const float* __restrict__ bias)
{
  const int tid  = threadIdx.x;
  const int lane = tid & 31;
  const int wave = tid >> 5;
  const int half = lane >> 4;
  const int l16  = lane & 15;

  // Phase 1: attention scores as WMMA  S[48,16] = s_x[48,256] @ pT^T
  // waves 0..2 each own one 16-row M-tile; EXEC stays all-ones per wave.
  if (wave < 3) {
    const int mt = wave;
    v8f sc;
    #pragma unroll
    for (int e = 0; e < 8; ++e) sc[e] = 0.f;
    for (int kt = 0; kt < 8; ++kt) {
      int k0 = kt * 32 + half * 8;
      Frag a, b;
      int m = mt * 16 + l16;
      a.u[0] = *(const uint4*)&s_x[m][k0];
      a.u[1] = *(const uint4*)&s_x[m][k0 + 16];
      const bf16* bp = pT + l16 * 256 + k0;
      b.u[0] = *(const uint4*)bp;
      b.u[1] = *(const uint4*)(bp + 16);
      sc = __builtin_amdgcn_wmma_f32_16x16x32_bf16(
          false, a.v, false, b.v, (short)0, sc, false, false);
    }
    if (l16 < 8) {                  // columns 0..7 hold the 8 score vectors
      #pragma unroll
      for (int c = 0; c < 8; ++c) {
        int row = mt * 16 + half * 8 + c;
        int smp = row / 3, node = row - smp * 3;
        s_sc[smp][l16 >> 2][node][l16 & 3] = sc[c];
      }
    }
  }
  __syncthreads();

  // Phase 2a: leaky-relu + softmax over sources j (3-wide)
  if (tid < TS * 12) {
    int h = tid & 3, i = (tid >> 2) % 3, smp = tid / 12;
    float sd = s_sc[smp][1][i][h];
    float e[3], m = -1e30f;
    #pragma unroll
    for (int j = 0; j < 3; ++j) {
      float v = sd + s_sc[smp][0][j][h];
      v = v >= 0.f ? v : 0.2f * v;
      e[j] = v; m = fmaxf(m, v);
    }
    float sum = 0.f;
    #pragma unroll
    for (int j = 0; j < 3; ++j) { e[j] = __expf(e[j] - m); sum += e[j]; }
    float inv = 1.f / sum;
    #pragma unroll
    for (int j = 0; j < 3; ++j) s_al[smp][i][j][h] = e[j] * inv;
  }
  __syncthreads();

  // Phase 2b: build mixed A-matrix  s_a[3s+i][h*256+f] = sum_j alpha[i,j,h]*x[3s+j][f]
  for (int c = tid; c < MROWS * 128; c += 256) {
    int r  = c >> 7;
    int kk = (c & 127) * 8;
    int h = kk >> 8, f0 = kk & 255;
    int smp = r / 3, i = r - smp * 3;
    float a0 = s_al[smp][i][0][h], a1 = s_al[smp][i][1][h], a2 = s_al[smp][i][2][h];
    B8 v0, v1, v2, o;
    v0.u = *(const uint4*)&s_x[smp * 3 + 0][f0];
    v1.u = *(const uint4*)&s_x[smp * 3 + 1][f0];
    v2.u = *(const uint4*)&s_x[smp * 3 + 2][f0];
    #pragma unroll
    for (int e = 0; e < 8; ++e)
      o.h[e] = (bf16)(a0 * (float)v0.h[e] + a1 * (float)v1.h[e] + a2 * (float)v2.h[e]);
    *(uint4*)&s_a[r][kk] = o.u;
  }
  __syncthreads();

  // Phase 3: WMMA GEMM  C[48,256] = s_a[48,1024] @ WrT^T, then relu(0.25*C + bias) -> s_x
  v8f acc[3][2];
  #pragma unroll
  for (int mt = 0; mt < 3; ++mt)
    #pragma unroll
    for (int j = 0; j < 2; ++j)
      #pragma unroll
      for (int e = 0; e < 8; ++e) acc[mt][j][e] = 0.f;

  const int nbase = wave * 32;            // each wave owns 2 N-tiles

  for (int kt = 0; kt < 32; ++kt) {
    int k0 = kt * 32 + half * 8;
    Frag a[3], b[2];
    #pragma unroll
    for (int mt = 0; mt < 3; ++mt) {
      int m = mt * 16 + l16;
      a[mt].u[0] = *(const uint4*)&s_a[m][k0];
      a[mt].u[1] = *(const uint4*)&s_a[m][k0 + 16];
    }
    #pragma unroll
    for (int j = 0; j < 2; ++j) {
      int n = nbase + j * 16 + l16;
      const bf16* bp = wrT + n * 1024 + k0;
      b[j].u[0] = *(const uint4*)bp;
      b[j].u[1] = *(const uint4*)(bp + 16);
    }
    #pragma unroll
    for (int mt = 0; mt < 3; ++mt)
      #pragma unroll
      for (int j = 0; j < 2; ++j)
        acc[mt][j] = __builtin_amdgcn_wmma_f32_16x16x32_bf16(
            false, a[mt].v, false, b[j].v, (short)0, acc[mt][j], false, false);
  }

  // epilogue: head-mean (0.25) + bias + relu, write bf16 y back into s_x
  #pragma unroll
  for (int mt = 0; mt < 3; ++mt)
    #pragma unroll
    for (int j = 0; j < 2; ++j) {
      int n = nbase + j * 16 + l16;
      float bc = bias[n];
      #pragma unroll
      for (int c2 = 0; c2 < 8; ++c2) {
        int row = mt * 16 + half * 8 + c2;
        float v = acc[mt][j][c2] * 0.25f + bc;
        s_x[row][n] = (bf16)fmaxf(v, 0.f);
      }
    }
  __syncthreads();
}

// ---------------- main fused kernel ----------------

__global__ __launch_bounds__(256) void gat_main(
    const float* __restrict__ text, const float* __restrict__ audio,
    const float* __restrict__ video, const float* __restrict__ type_emb,
    const bf16* __restrict__ pT1, const bf16* __restrict__ wrT1, const float* __restrict__ b1,
    const bf16* __restrict__ pT2, const bf16* __restrict__ wrT2, const float* __restrict__ b2,
    const bf16* __restrict__ woT, const float* __restrict__ bo,
    float* __restrict__ out)
{
  __shared__ bf16  s_x[MROWS][256];        // 24 KB  (x, then y1, then y2)
  __shared__ bf16  s_a[MROWS][1024];       // 96 KB  (alpha-mixed A matrix)
  __shared__ float s_sc[TS][2][3][4];      // 1.5 KB scores
  __shared__ float s_al[TS][3][3][4];      // 2.25 KB alpha
  __shared__ bf16  s_pool[TS][256];        // 8 KB   pooled

  const int tid = threadIdx.x;
  const int b0  = blockIdx.x * TS;

  // Phase 0: stage x + type_emb as bf16
  const float* feats[3] = {text, audio, video};
  for (int c = tid; c < MROWS * 64; c += 256) {
    int r  = c >> 6;
    int f0 = (c & 63) * 4;
    int smp = r / 3, node = r - smp * 3;
    const float4 xv = *(const float4*)(feats[node] + (size_t)(b0 + smp) * 256 + f0);
    const float4 tv = *(const float4*)(type_emb + node * 256 + f0);
    B4 o;
    o.h[0] = (bf16)(xv.x + tv.x); o.h[1] = (bf16)(xv.y + tv.y);
    o.h[2] = (bf16)(xv.z + tv.z); o.h[3] = (bf16)(xv.w + tv.w);
    *(uint2*)&s_x[r][f0] = o.u;
  }
  __syncthreads();

  gat_layer(s_x, s_a, s_sc, s_al, pT1, wrT1, b1);
  gat_layer(s_x, s_a, s_sc, s_al, pT2, wrT2, b2);

  // pool over the 3 nodes
  for (int c = tid; c < TS * 32; c += 256) {
    int smp = c >> 5;
    int f0  = (c & 31) * 8;
    B8 y0, y1, y2, o;
    y0.u = *(const uint4*)&s_x[smp * 3 + 0][f0];
    y1.u = *(const uint4*)&s_x[smp * 3 + 1][f0];
    y2.u = *(const uint4*)&s_x[smp * 3 + 2][f0];
    #pragma unroll
    for (int e = 0; e < 8; ++e) {
      float v = ((float)y0.h[e] + (float)y1.h[e] + (float)y2.h[e]) * (1.f / 3.f);
      o.h[e] = (bf16)v;
    }
    *(uint4*)&s_pool[smp][f0] = o.u;
  }
  __syncthreads();

  // final GEMM: out[b0+s, n] = pooled[s,:] @ Wo + bo   (M=16, K=256, N=256)
  const int lane = tid & 31, wave = tid >> 5;
  const int half = lane >> 4, l16 = lane & 15;
  const int nbase = wave * 32;
  v8f acc[2];
  #pragma unroll
  for (int j = 0; j < 2; ++j)
    #pragma unroll
    for (int e = 0; e < 8; ++e) acc[j][e] = 0.f;

  for (int kt = 0; kt < 8; ++kt) {
    int k0 = kt * 32 + half * 8;
    Frag a, b[2];
    a.u[0] = *(const uint4*)&s_pool[l16][k0];
    a.u[1] = *(const uint4*)&s_pool[l16][k0 + 16];
    #pragma unroll
    for (int j = 0; j < 2; ++j) {
      int n = nbase + j * 16 + l16;
      const bf16* bp = woT + n * 256 + k0;
      b[j].u[0] = *(const uint4*)bp;
      b[j].u[1] = *(const uint4*)(bp + 16);
    }
    #pragma unroll
    for (int j = 0; j < 2; ++j)
      acc[j] = __builtin_amdgcn_wmma_f32_16x16x32_bf16(
          false, a.v, false, b[j].v, (short)0, acc[j], false, false);
  }
  #pragma unroll
  for (int j = 0; j < 2; ++j) {
    int n = nbase + j * 16 + l16;
    float bb = bo[n];
    #pragma unroll
    for (int c2 = 0; c2 < 8; ++c2) {
      int smp = half * 8 + c2;
      out[(size_t)(b0 + smp) * 256 + n] = acc[j][c2] + bb;
    }
  }
}

// ---------------- host-side launch ----------------

extern "C" void kernel_launch(void* const* d_in, const int* in_sizes, int n_in,
                              void* d_out, int out_size, void* d_ws, size_t ws_size,
                              hipStream_t stream) {
  (void)in_sizes; (void)n_in; (void)out_size; (void)ws_size;
  const float* text      = (const float*)d_in[0];
  const float* audio     = (const float*)d_in[1];
  const float* video     = (const float*)d_in[2];
  const float* type_emb  = (const float*)d_in[3];
  const float* W1        = (const float*)d_in[4];
  const float* a_src1    = (const float*)d_in[5];
  const float* a_dst1    = (const float*)d_in[6];
  const float* b1        = (const float*)d_in[7];
  const float* W2        = (const float*)d_in[8];
  const float* a_src2    = (const float*)d_in[9];
  const float* a_dst2    = (const float*)d_in[10];
  const float* b2        = (const float*)d_in[11];
  const float* Wo        = (const float*)d_in[12];
  const float* bo        = (const float*)d_in[13];

  char* ws = (char*)d_ws;
  bf16* wrT1 = (bf16*)(ws + 0);            // 512 KB
  bf16* wrT2 = (bf16*)(ws + 524288);       // 512 KB
  bf16* woT  = (bf16*)(ws + 1048576);      // 128 KB
  bf16* pT1  = (bf16*)(ws + 1179648);      // 8 KB  [16][256] bf16
  bf16* pT2  = (bf16*)(ws + 1187840);      // 8 KB

  prep_wr<<<256, 256, 0, stream>>>(W1, wrT1);
  prep_wr<<<256, 256, 0, stream>>>(W2, wrT2);
  prep_wo<<<256, 256, 0, stream>>>(Wo, woT);
  prep_p<<<16, 256, 0, stream>>>(W1, a_src1, a_dst1, pT1);
  prep_p<<<16, 256, 0, stream>>>(W2, a_src2, a_dst2, pT2);

  gat_main<<<B_TOTAL / TS, 256, 0, stream>>>(
      text, audio, video, type_emb,
      pT1, wrT1, b1, pT2, wrT2, b2, woT, bo, (float*)d_out);
}
